// AttentionalGNN_60498909331815
// MI455X (gfx1250) — matrix operand order repaired and checked by reading, not compiled, not run on hardware
//
#include <hip/hip_runtime.h>

// ---------------------------------------------------------------------------
// Constants matching the reference
// ---------------------------------------------------------------------------
constexpr int cB     = 8;
constexpr int cC     = 256;            // D_MODEL
constexpr int cN     = 1024;
constexpr int cH     = 4;
constexpr int cDH    = 64;             // HEAD_DIM
constexpr int cG     = 64;             // GNN_DIM
constexpr int cKNN   = 8;
constexpr int cKT    = 25;             // KSIZE*KSIZE
constexpr int cNODES = cB * cN;        // 8192
constexpr int cE     = cB * cN * cKNN; // 65536
constexpr int cSTEPS = 10;

typedef __attribute__((ext_vector_type(16))) _Float16 v16h;
typedef __attribute__((ext_vector_type(8)))  _Float16 v8h;
typedef __attribute__((ext_vector_type(8)))  float    v8f;

union Frag16 { v16h v; v8h p[2]; _Float16 h[16]; };
union Oct16  { v8h  v; _Float16 h[8]; };
union CFrag  { v8f  v; float    f[8]; };

// ---------------------------------------------------------------------------
// General batched WMMA GEMM:  C = act( scale * A(MxK) * B(KxN) + bias )
// REQUIRES: M % 128 == 0, N % 64 == 0, K % 32 == 0 (true for every call site),
// lda/ldb multiples of 8 floats (16B-aligned rows).
// A row-major (lda); B row-major (ldb), or B is (N x K) row-major when transB.
// Per-z batch offsets sA/sB/sC in elements.
// Tile 128x64, 4 waves; each wave owns two 16-row strips and reuses each B
// fragment across both (12 ds_load_b128 per 8 WMMA per K-step).  Tiles are
// staged into LDS in WMMA fragment order so fragments load as ds_load_b128.
// ---------------------------------------------------------------------------
__global__ __launch_bounds__(128) void gemm_wmma(
    const float* __restrict__ A, const float* __restrict__ Bm,
    const float* __restrict__ bias, float* __restrict__ Cm,
    int M_, int N_, int K_, int lda, int ldb, int ldc,
    long long sA, long long sB, long long sC,
    int transB, int act, float scale)
{
  // [strip][lane][e] : lane's 16 fragment halves are contiguous (32B)
  __shared__ _Float16 lAf[8][32][16];
  // [ntile][lane][e]
  __shared__ _Float16 lBf[4][32][16];

  A  += (long long)blockIdx.z * sA;
  Bm += (long long)blockIdx.z * sB;
  Cm += (long long)blockIdx.z * sC;
  const int row0 = blockIdx.y * 128;
  const int col0 = blockIdx.x * 64;
  const int t    = threadIdx.x;
  const int lane = t & 31;
  const int wave = t >> 5;
  const int half = lane >> 4;
  const int l16  = lane & 15;

  CFrag acc[2][4];
#pragma unroll
  for (int p = 0; p < 2; ++p)
#pragma unroll
    for (int i = 0; i < 4; ++i)
#pragma unroll
      for (int j = 0; j < 8; ++j) acc[p][i].f[j] = 0.0f;

  for (int kb = 0; kb < K_; kb += 32) {
    // ---- stage A tile (128 rows x 32 k) into fragment order ----
    // task = (row, k-octet); 8 consecutive k for one row -> 8 contiguous e.
#pragma unroll
    for (int task = t; task < 512; task += 128) {
      const int r  = task >> 2;          // 0..127
      const int oc = (task & 3) << 3;    // 0,8,16,24
      const float4* ap =
          (const float4*)(A + (long long)(row0 + r) * lda + kb + oc);
      float4 x0 = ap[0], x1 = ap[1];
      Oct16 o;
      o.h[0] = (_Float16)x0.x; o.h[1] = (_Float16)x0.y;
      o.h[2] = (_Float16)x0.z; o.h[3] = (_Float16)x0.w;
      o.h[4] = (_Float16)x1.x; o.h[5] = (_Float16)x1.y;
      o.h[6] = (_Float16)x1.z; o.h[7] = (_Float16)x1.w;
      const int hf = (oc >> 4) & 1;          // k-half (lane group)
      const int e0 = oc & 8;                 // 0 or 8
      *(v8h*)&lAf[r >> 4][hf * 16 + (r & 15)][e0] = o.v;
    }
    // ---- stage B tile (32 k x 64 n) into fragment order ----
    if (transB) {
      // B is N x K: 8 consecutive k contiguous in global AND in LDS.
#pragma unroll
      for (int task = t; task < 256; task += 128) {
        const int n  = task & 63;
        const int ko = (task >> 6) << 3;     // 0,8,16,24
        const float4* bp =
            (const float4*)(Bm + (long long)(col0 + n) * ldb + kb + ko);
        float4 x0 = bp[0], x1 = bp[1];
        Oct16 o;
        o.h[0] = (_Float16)x0.x; o.h[1] = (_Float16)x0.y;
        o.h[2] = (_Float16)x0.z; o.h[3] = (_Float16)x0.w;
        o.h[4] = (_Float16)x1.x; o.h[5] = (_Float16)x1.y;
        o.h[6] = (_Float16)x1.z; o.h[7] = (_Float16)x1.w;
        const int ln = ((ko >> 4) & 1) * 16 + (n & 15);
        *(v8h*)&lBf[n >> 4][ln][ko & 8] = o.v;
      }
    } else {
      // B is K x N: keep global loads n-coalesced; scatter halves to LDS.
#pragma unroll
      for (int task = t; task < 256; task += 128) {
        const int n  = task & 63;
        const int ko = (task >> 6) << 3;
        const int ln = ((ko >> 4) & 1) * 16 + (n & 15);
        const int nt = n >> 4;
        const int e0 = ko & 8;
        const float* bp = Bm + (long long)(kb + ko) * ldb + col0 + n;
#pragma unroll
        for (int j = 0; j < 8; ++j)
          lBf[nt][ln][e0 + j] = (_Float16)bp[(long long)j * ldb];
      }
    }
    __syncthreads();

    Frag16 af0, af1;
    af0.p[0] = *(const v8h*)&lAf[wave][lane][0];
    af0.p[1] = *(const v8h*)&lAf[wave][lane][8];
    af1.p[0] = *(const v8h*)&lAf[4 + wave][lane][0];
    af1.p[1] = *(const v8h*)&lAf[4 + wave][lane][8];
#pragma unroll
    for (int nt = 0; nt < 4; ++nt) {
      Frag16 bf;
      bf.p[0] = *(const v8h*)&lBf[nt][lane][0];
      bf.p[1] = *(const v8h*)&lBf[nt][lane][8];
      acc[0][nt].v = __builtin_amdgcn_wmma_f32_16x16x32_f16(
          false, af0.v, false, bf.v, (short)0, acc[0][nt].v, false, false);
      acc[1][nt].v = __builtin_amdgcn_wmma_f32_16x16x32_f16(
          false, af1.v, false, bf.v, (short)0, acc[1][nt].v, false, false);
    }
    __syncthreads();
  }

  // C layout: VGPR r -> M = r + 8*half, N = lane&15
#pragma unroll
  for (int p = 0; p < 2; ++p) {
    const int rbase = row0 + p * 64 + wave * 16 + 8 * half;
#pragma unroll
    for (int nt = 0; nt < 4; ++nt) {
      const int gc = col0 + nt * 16 + l16;
#pragma unroll
      for (int r = 0; r < 8; ++r) {
        float v = acc[p][nt].f[r] * scale;
        if (bias) v += bias[gc];
        if (act == 1) v = fmaxf(v, 0.0f);
        else if (act == 2) v = 0.5f * v * (1.0f + erff(v * 0.70710678118654752f));
        Cm[(long long)(rbase + r) * ldc + gc] = v;
      }
    }
  }
}

// ---------------------------------------------------------------------------
// LayerNorm over contiguous rows (blockDim.x == Cc, Cc in {64,256})
// ---------------------------------------------------------------------------
__global__ void ln_rows(const float* __restrict__ X, const float* __restrict__ g,
                        const float* __restrict__ b, float* __restrict__ Y, int Cc)
{
  __shared__ float red[256];
  const int r = blockIdx.x, t = threadIdx.x;
  float v = X[(size_t)r * Cc + t];
  red[t] = v; __syncthreads();
  for (int s = blockDim.x >> 1; s > 0; s >>= 1) { if (t < s) red[t] += red[t + s]; __syncthreads(); }
  float mean = red[0] / (float)Cc; __syncthreads();
  float d = v - mean;
  red[t] = d * d; __syncthreads();
  for (int s = blockDim.x >> 1; s > 0; s >>= 1) { if (t < s) red[t] += red[t + s]; __syncthreads(); }
  float var = red[0] / (float)Cc;
  Y[(size_t)r * Cc + t] = d * rsqrtf(var + 1e-5f) * g[t] + b[t];
}

// LayerNorm reading desc in (B,C,N) layout, writing (B*N, C) rows.
__global__ __launch_bounds__(256) void ln_bcn(const float* __restrict__ D,
                                              const float* __restrict__ g,
                                              const float* __restrict__ b,
                                              float* __restrict__ Y)
{
  __shared__ float red[256];
  const int r = blockIdx.x, t = threadIdx.x;
  const int bb = r / cN, n = r % cN;
  float v = D[(size_t)bb * cC * cN + (size_t)t * cN + n];
  red[t] = v; __syncthreads();
  for (int s = 128; s > 0; s >>= 1) { if (t < s) red[t] += red[t + s]; __syncthreads(); }
  float mean = red[0] / (float)cC; __syncthreads();
  float d = v - mean;
  red[t] = d * d; __syncthreads();
  for (int s = 128; s > 0; s >>= 1) { if (t < s) red[t] += red[t + s]; __syncthreads(); }
  float var = red[0] / (float)cC;
  Y[(size_t)r * cC + t] = d * rsqrtf(var + 1e-5f) * g[t] + b[t];
}

// ---------------------------------------------------------------------------
// Row softmax over S (per head h = blockIdx.y, row = blockIdx.x), optional
// accumulation of mean-over-heads attention into awm (per batch base ptr).
// ---------------------------------------------------------------------------
__global__ __launch_bounds__(256) void softmax_awm(float* __restrict__ S,
                                                   float* __restrict__ awmb,
                                                   float invH)
{
  __shared__ float red[256];
  const int row = blockIdx.x, h = blockIdx.y, t = threadIdx.x;
  float* Sr = S + ((size_t)h * cN + row) * cN;
  float mx = -3.4e38f;
  for (int m = t; m < cN; m += 256) mx = fmaxf(mx, Sr[m]);
  red[t] = mx; __syncthreads();
  for (int s = 128; s > 0; s >>= 1) { if (t < s) red[t] = fmaxf(red[t], red[t + s]); __syncthreads(); }
  mx = red[0]; __syncthreads();
  float sum = 0.0f;
  for (int m = t; m < cN; m += 256) { float e = __expf(Sr[m] - mx); Sr[m] = e; sum += e; }
  red[t] = sum; __syncthreads();
  for (int s = 128; s > 0; s >>= 1) { if (t < s) red[t] += red[t + s]; __syncthreads(); }
  float inv = 1.0f / red[0];
  if (awmb) {
    float* ar = awmb + (size_t)row * cN;
    for (int m = t; m < cN; m += 256) { float a = Sr[m] * inv; Sr[m] = a; atomicAdd(&ar[m], a * invH); }
  } else {
    for (int m = t; m < cN; m += 256) Sr[m] *= inv;
  }
}

// ---------------------------------------------------------------------------
// Small utility kernels
// ---------------------------------------------------------------------------
__global__ void zero_f(float* p, long long n)
{ for (long long i = blockIdx.x * (long long)blockDim.x + threadIdx.x; i < n; i += (long long)gridDim.x * blockDim.x) p[i] = 0.0f; }

__global__ void copy_f(float* dst, const float* src, long long n)
{ for (long long i = blockIdx.x * (long long)blockDim.x + threadIdx.x; i < n; i += (long long)gridDim.x * blockDim.x) dst[i] = src[i]; }

__global__ void add_inplace(float* y, const float* x, long long n)
{ for (long long i = blockIdx.x * (long long)blockDim.x + threadIdx.x; i < n; i += (long long)gridDim.x * blockDim.x) y[i] += x[i]; }

__global__ __launch_bounds__(512) void concat2_k(const float* a, const float* b, float* out)
{ int r = blockIdx.x, t = threadIdx.x;
  out[(size_t)r * 512 + t] = (t < 256) ? a[(size_t)r * 256 + t] : b[(size_t)r * 256 + (t - 256)]; }

__global__ __launch_bounds__(192) void concat3_k(const float* a, const float* b, const float* c, float* out)
{ int r = blockIdx.x, t = threadIdx.x;
  float v = (t < 64) ? a[(size_t)r * 64 + t] : (t < 128 ? b[(size_t)r * 64 + (t - 64)] : c[(size_t)r * 64 + (t - 128)]);
  out[(size_t)r * 192 + t] = v; }

// cat0[b][n][:] = [ delta[b*N+n][0:256] | hln[n*B+b][0:64] ]
__global__ __launch_bounds__(320) void cat_delta_h(const float* delta, const float* hln, float* out)
{ int r = blockIdx.x, t = threadIdx.x;
  int b = r / cN, n = r % cN;
  float v = (t < 256) ? delta[(size_t)r * 256 + t]
                      : hln[((size_t)n * cB + b) * 64 + (t - 256)];
  out[(size_t)r * 320 + t] = v; }

// desc(B,C,N) += m(B*N, C)^T
__global__ __launch_bounds__(256) void add_transpose_k(float* desc, const float* m)
{ int r = blockIdx.x, c = threadIdx.x;
  int b = r / cN, n = r % cN;
  desc[(size_t)b * cC * cN + (size_t)c * cN + n] += m[(size_t)r * cC + c]; }

// x_s[n*B+b][c] = desc[b][c][n]
__global__ __launch_bounds__(256) void desc_to_nodes(const float* desc, float* xs)
{ int r = blockIdx.x, c = threadIdx.x;
  int b = r / cN, n = r % cN;
  xs[((size_t)n * cB + b) * cC + c] = desc[(size_t)b * cC * cN + (size_t)c * cN + n]; }

// ---------------------------------------------------------------------------
// top-8 per row of awm (one thread per (b,n) row)
// ---------------------------------------------------------------------------
__global__ void topk8_kernel(const float* __restrict__ awm, int* __restrict__ out)
{
  int r = blockIdx.x * blockDim.x + threadIdx.x;
  if (r >= cB * cN) return;
  const float* a = awm + (size_t)r * cN;
  int ch[cKNN];
#pragma unroll
  for (int j = 0; j < cKNN; ++j) {
    float best = -3.4e38f; int bi = 0;
    for (int m = 0; m < cN; ++m) {
      bool used = false;
      for (int u = 0; u < j; ++u) used |= (ch[u] == m);
      if (!used && a[m] > best) { best = a[m]; bi = m; }
    }
    ch[j] = bi;
    out[(size_t)r * cKNN + j] = bi;
  }
}

// Replicates the reference's flat-index quirk: indices address kpts rows <1024.
__global__ void edge_build(const int* topk, const float* kpts, int* src, int* dst, float* attr)
{
  int e = blockIdx.x * blockDim.x + threadIdx.x;
  if (e >= cE) return;
  int i0 = topk[e];
  int n  = (e / cKNN) % cN;
  src[e] = i0; dst[e] = n;
  attr[e * 2 + 0] = kpts[i0 * 2 + 0] - kpts[n * 2 + 0];
  attr[e * 2 + 1] = kpts[i0 * 2 + 1] - kpts[n * 2 + 1];
}

__global__ __launch_bounds__(256) void minmax_attr(const float* attr, float* mm)
{
  __shared__ float s0[256], s1[256], s2[256], s3[256];
  int t = threadIdx.x;
  float mn0 = 3.4e38f, mn1 = 3.4e38f, mx0 = -3.4e38f, mx1 = -3.4e38f;
  for (int e = t; e < cE; e += 256) {
    float a0 = attr[e * 2], a1 = attr[e * 2 + 1];
    mn0 = fminf(mn0, a0); mx0 = fmaxf(mx0, a0);
    mn1 = fminf(mn1, a1); mx1 = fmaxf(mx1, a1);
  }
  s0[t] = mn0; s1[t] = mn1; s2[t] = mx0; s3[t] = mx1; __syncthreads();
  for (int s = 128; s > 0; s >>= 1) {
    if (t < s) { s0[t] = fminf(s0[t], s0[t+s]); s1[t] = fminf(s1[t], s1[t+s]);
                 s2[t] = fmaxf(s2[t], s2[t+s]); s3[t] = fmaxf(s3[t], s3[t+s]); }
    __syncthreads();
  }
  if (t == 0) { mm[0] = s0[0]; mm[1] = s1[0]; mm[2] = s2[0]; mm[3] = s3[0]; }
}

__global__ void basis_kernel(const float* attr, const float* mm, int* kidx, float* bw)
{
  int e = blockIdx.x * blockDim.x + threadIdx.x;
  if (e >= cE) return;
  float a0 = (attr[e * 2 + 0] - mm[0]) / (mm[2] - mm[0]);
  float a1 = (attr[e * 2 + 1] - mm[1]) / (mm[3] - mm[1]);
  float p0 = a0 * 4.0f, p1 = a1 * 4.0f;
  float l0 = fminf(fmaxf(floorf(p0), 0.0f), 3.0f);
  float l1 = fminf(fmaxf(floorf(p1), 0.0f), 3.0f);
  float f0 = p0 - l0, f1 = p1 - l1;
  int i0 = (int)l0, i1 = (int)l1;
  const int c0s[4] = {0, 0, 1, 1}, c1s[4] = {0, 1, 0, 1};
#pragma unroll
  for (int j = 0; j < 4; ++j) {
    kidx[(size_t)e * 4 + j] = (i0 + c0s[j]) * 5 + (i1 + c1s[j]);
    float w0 = c0s[j] ? f0 : 1.0f - f0;
    float w1 = c1s[j] ? f1 : 1.0f - f1;
    bw[(size_t)e * 4 + j] = w0 * w1;
  }
}

// msg[e][d] = sum_j bw[e][j]*hk[src[e]][kidx[e][j]][d]; scatter into agg/deg.
__global__ __launch_bounds__(64) void spline_msg(const float* __restrict__ hk,
                                                 const int* __restrict__ src,
                                                 const int* __restrict__ dst,
                                                 const int* __restrict__ kidx,
                                                 const float* __restrict__ bw,
                                                 float* __restrict__ agg,
                                                 float* __restrict__ deg)
{
  int e = blockIdx.x, d = threadIdx.x;
  const float* hks = hk + (size_t)src[e] * (cKT * cG);
  float acc = 0.0f;
#pragma unroll
  for (int j = 0; j < 4; ++j)
    acc += bw[(size_t)e * 4 + j] * hks[kidx[(size_t)e * 4 + j] * cG + d];
  atomicAdd(&agg[(size_t)dst[e] * cG + d], acc);
  if (d == 0) atomicAdd(&deg[dst[e]], 1.0f);
}

// out = relu( agg/max(deg,1) + xroot ), xroot already includes conv bias.
__global__ __launch_bounds__(64) void combine_conv(const float* agg, const float* deg,
                                                   const float* xroot, float* out)
{
  int n = blockIdx.x, d = threadIdx.x;
  float v = agg[(size_t)n * cG + d] / fmaxf(deg[n], 1.0f) + xroot[(size_t)n * cG + d];
  out[(size_t)n * cG + d] = fmaxf(v, 0.0f);
}

// ---------------------------------------------------------------------------
// Host-side orchestration
// ---------------------------------------------------------------------------
struct LayerP {
  const float *k_b,*k_w,*merge_b,*merge_w,*mlp1_b,*mlp1_w,*mlp2_b,*mlp2_w;
  const float *n0b,*n0g,*n1b,*n1g,*n2b,*n2g,*q_b,*q_w,*v_b,*v_w;
};
struct DgmcP {
  const float *c1_W,*c1_b,*c1_root,*c2_W,*c2_b,*c2_root,*lin_b,*lin_w,*proj_b,*proj_w;
};
struct WSpace {
  float *desc0,*desc1,*xln,*srcln,*q,*k,*v,*S,*awm,*msg,*t0,*t1,*cat512,*m1o;
  float *deltaA,*deltaB,*xs,*h,*h1,*h2,*hk,*agg,*deg,*xroot,*cat192,*gout,*hln,*m0;
  float *attr,*bw,*mm;
  int *topk,*src,*dst,*kidx;
};

static void run_gemm(hipStream_t st, const float* A, const float* Bm, const float* bias,
                     float* Cm, int M, int Nn, int K, int lda, int ldb, int ldc,
                     long long sA, long long sB, long long sC, int nz,
                     int transB, int act, float scale)
{
  dim3 grid((Nn + 63) / 64, (M + 127) / 128, nz);
  gemm_wmma<<<grid, dim3(128), 0, st>>>(A, Bm, bias, Cm, M, Nn, K, lda, ldb, ldc,
                                        sA, sB, sC, transB, act, scale);
}

static void encoder_run(hipStream_t st, const float* xdesc, const float* sdesc,
                        const LayerP& L, float* delta, float* awm, WSpace& w)
{
  ln_bcn<<<cB * cN, 256, 0, st>>>(xdesc, L.n0g, L.n0b, w.xln);
  const float* sln = w.xln;
  if (sdesc != xdesc) {
    ln_bcn<<<cB * cN, 256, 0, st>>>(sdesc, L.n0g, L.n0b, w.srcln);
    sln = w.srcln;
  }
  run_gemm(st, w.xln, L.q_w, L.q_b, w.q, cNODES, cC, cC, cC, cC, cC, 0, 0, 0, 1, 0, 0, 1.0f);
  run_gemm(st, sln,   L.k_w, L.k_b, w.k, cNODES, cC, cC, cC, cC, cC, 0, 0, 0, 1, 0, 0, 1.0f);
  run_gemm(st, sln,   L.v_w, L.v_b, w.v, cNODES, cC, cC, cC, cC, cC, 0, 0, 0, 1, 0, 0, 1.0f);
  if (awm) zero_f<<<1024, 256, 0, st>>>(awm, (long long)cB * cN * cN);

  for (int b = 0; b < cB; ++b) {
    const float* qb = w.q + (size_t)b * cN * cC;
    const float* kb = w.k + (size_t)b * cN * cC;
    const float* vb = w.v + (size_t)b * cN * cC;
    // S[h] = (Q_h K_h^T)/8 : per head via z-stride 64 into the packed heads
    run_gemm(st, qb, kb, nullptr, w.S, cN, cN, cDH, cC, cC, cN,
             64, 64, (long long)cN * cN, cH, 1, 0, 0.125f);
    softmax_awm<<<dim3(cN, cH), 256, 0, st>>>(w.S, awm ? awm + (size_t)b * cN * cN : nullptr,
                                              1.0f / cH);
    // msg_h = A_h V_h, written back into packed-head layout
    run_gemm(st, w.S, vb, nullptr, w.msg + (size_t)b * cN * cC, cN, cDH, cN, cN, cC, cC,
             (long long)cN * cN, 64, 64, cH, 0, 0, 1.0f);
  }
  run_gemm(st, w.msg, L.merge_w, L.merge_b, w.t0, cNODES, cC, cC, cC, cC, cC, 0, 0, 0, 1, 0, 0, 1.0f);
  ln_rows<<<cNODES, 256, 0, st>>>(w.t0, L.n1g, L.n1b, w.t1, cC);
  concat2_k<<<cNODES, 512, 0, st>>>(w.xln, w.t1, w.cat512);
  run_gemm(st, w.cat512, L.mlp1_w, L.mlp1_b, w.m1o, cNODES, 512, 512, 512, 512, 512, 0, 0, 0, 1, 0, 2, 1.0f);
  run_gemm(st, w.m1o, L.mlp2_w, L.mlp2_b, delta, cNODES, cC, 512, 512, cC, cC, 0, 0, 0, 1, 0, 0, 1.0f);
  ln_rows<<<cNODES, 256, 0, st>>>(delta, L.n2g, L.n2b, delta, cC);
}

static void spline_conv_run(hipStream_t st, const float* x, const float* W,
                            const float* root, const float* bias, WSpace& w, float* out)
{
  // hk[n][k][d] = x[n][:] @ W[k]  (batched over 25 taps)
  run_gemm(st, x, W, nullptr, w.hk, cNODES, cG, cG, cG, cG, cKT * cG,
           0, (long long)cG * cG, cG, cKT, 0, 0, 1.0f);
  run_gemm(st, x, root, bias, w.xroot, cNODES, cG, cG, cG, cG, cG, 0, 0, 0, 1, 0, 0, 1.0f);
  zero_f<<<1024, 256, 0, st>>>(w.agg, (long long)cNODES * cG);
  zero_f<<<32, 256, 0, st>>>(w.deg, (long long)cNODES);
  spline_msg<<<cE, 64, 0, st>>>(w.hk, w.src, w.dst, w.kidx, w.bw, w.agg, w.deg);
  combine_conv<<<cNODES, 64, 0, st>>>(w.agg, w.deg, w.xroot, out);
}

static void self_tail_run(hipStream_t st, float* descb, const float* kpts,
                          const DgmcP& dg, const float* m1_w, const float* m1_b,
                          const float* m2_w, const float* m2_b,
                          const float* gn1g, const float* gn1b, WSpace& w)
{
  topk8_kernel<<<(cB * cN + 255) / 256, 256, 0, st>>>(w.awm, w.topk);
  edge_build<<<(cE + 255) / 256, 256, 0, st>>>(w.topk, kpts, w.src, w.dst, w.attr);
  minmax_attr<<<1, 256, 0, st>>>(w.attr, w.mm);
  basis_kernel<<<(cE + 255) / 256, 256, 0, st>>>(w.attr, w.mm, w.kidx, w.bw);
  desc_to_nodes<<<cB * cN, 256, 0, st>>>(descb, w.xs);
  // h = relu(x_s @ proj)
  run_gemm(st, w.xs, dg.proj_w, dg.proj_b, w.h, cNODES, cG, cC, cC, cG, cG, 0, 0, 0, 1, 0, 1, 1.0f);
  for (int step = 0; step < cSTEPS; ++step) {
    spline_conv_run(st, w.h,  dg.c1_W, dg.c1_root, dg.c1_b, w, w.h1);
    spline_conv_run(st, w.h1, dg.c2_W, dg.c2_root, dg.c2_b, w, w.h2);
    concat3_k<<<cNODES, 192, 0, st>>>(w.h, w.h1, w.h2, w.cat192);
    run_gemm(st, w.cat192, dg.lin_w, dg.lin_b, w.gout, cNODES, cG, 3 * cG,
             3 * cG, cG, cG, 0, 0, 0, 1, 0, 0, 1.0f);
    add_inplace<<<1024, 256, 0, st>>>(w.h, w.gout, (long long)cNODES * cG);
  }
  ln_rows<<<cNODES, 64, 0, st>>>(w.h, gn1g, gn1b, w.hln, cG);
  cat_delta_h<<<cNODES, 320, 0, st>>>(w.deltaA, w.hln, w.cat512 /*reused as cat320*/);
  run_gemm(st, w.cat512, m1_w, m1_b, w.m1o, cNODES, 320, 320, 320, 320, 320, 0, 0, 0, 1, 0, 2, 1.0f);
  run_gemm(st, w.m1o, m2_w, m2_b, w.m0, cNODES, cC, 320, 320, cC, cC, 0, 0, 0, 1, 0, 0, 1.0f);
  add_transpose_k<<<cB * cN, 256, 0, st>>>(descb, w.m0);
}

// ---------------------------------------------------------------------------
extern "C" void kernel_launch(void* const* d_in, const int* in_sizes, int n_in,
                              void* d_out, int out_size, void* d_ws, size_t ws_size,
                              hipStream_t stream)
{
  (void)in_sizes; (void)n_in; (void)out_size; (void)ws_size;
  const float* desc0_in = (const float*)d_in[0];
  const float* desc1_in = (const float*)d_in[1];
  const float* kpts0    = (const float*)d_in[2];
  const float* kpts1    = (const float*)d_in[3];

  // --- unpack params in JAX pytree (sorted-dict-key) leaf order ---
  int pi = 4;
  auto nx = [&]() -> const float* { return (const float*)d_in[pi++]; };
  DgmcP dg;
  dg.c1_W = nx(); dg.c1_b = nx(); dg.c1_root = nx();   // dgmc.conv1: W,b,root
  dg.c2_W = nx(); dg.c2_b = nx(); dg.c2_root = nx();   // dgmc.conv2
  dg.lin_b = nx(); dg.lin_w = nx();                    // dgmc.lin: b,w
  dg.proj_b = nx(); dg.proj_w = nx();                  // dgmc.proj: b,w
  LayerP Ls[4];
  for (int i = 0; i < 4; ++i) {
    LayerP& L = Ls[i];
    L.k_b = nx(); L.k_w = nx();
    L.merge_b = nx(); L.merge_w = nx();
    L.mlp1_b = nx(); L.mlp1_w = nx();
    L.mlp2_b = nx(); L.mlp2_w = nx();
    L.n0b = nx(); L.n0g = nx();
    L.n1b = nx(); L.n1g = nx();
    L.n2b = nx(); L.n2g = nx();
    L.q_b = nx(); L.q_w = nx();
    L.v_b = nx(); L.v_w = nx();
  }
  const float* m1_b = nx(); const float* m1_w = nx();
  const float* m2_b = nx(); const float* m2_w = nx();
  const float* gn1b = nx(); const float* gn1g = nx();

  // --- workspace carve-out ---
  char* ws = (char*)d_ws;
  size_t off = 0;
  auto allocf = [&](size_t n) -> float* { float* p = (float*)(ws + off); off += ((n * 4 + 255) / 256) * 256; return p; };
  auto alloci = [&](size_t n) -> int*   { int*   p = (int*)  (ws + off); off += ((n * 4 + 255) / 256) * 256; return p; };

  const size_t SD = (size_t)cB * cC * cN;     // 2,097,152
  WSpace w;
  w.desc0 = allocf(SD);            w.desc1 = allocf(SD);
  w.xln   = allocf((size_t)cNODES * cC);  w.srcln = allocf((size_t)cNODES * cC);
  w.q     = allocf((size_t)cNODES * cC);  w.k = allocf((size_t)cNODES * cC);  w.v = allocf((size_t)cNODES * cC);
  w.S     = allocf((size_t)cH * cN * cN);
  w.awm   = allocf((size_t)cB * cN * cN);
  w.msg   = allocf((size_t)cNODES * cC);
  w.t0    = allocf((size_t)cNODES * cC);  w.t1 = allocf((size_t)cNODES * cC);
  w.cat512 = allocf((size_t)cNODES * 512);
  w.m1o    = allocf((size_t)cNODES * 512);
  w.deltaA = allocf((size_t)cNODES * cC); w.deltaB = allocf((size_t)cNODES * cC);
  w.xs    = allocf((size_t)cNODES * cC);
  w.h     = allocf((size_t)cNODES * cG);
  w.h1    = allocf((size_t)cNODES * cG);  w.h2 = allocf((size_t)cNODES * cG);
  w.hk    = allocf((size_t)cNODES * cKT * cG);
  w.agg   = allocf((size_t)cNODES * cG);
  w.deg   = allocf((size_t)cNODES);
  w.xroot = allocf((size_t)cNODES * cG);
  w.cat192 = allocf((size_t)cNODES * 3 * cG);
  w.gout  = allocf((size_t)cNODES * cG);
  w.hln   = allocf((size_t)cNODES * cG);
  w.m0    = allocf((size_t)cNODES * cC);
  w.attr  = allocf((size_t)cE * 2);
  w.bw    = allocf((size_t)cE * 4);
  w.mm    = allocf(4);
  w.topk  = alloci((size_t)cE);
  w.src   = alloci((size_t)cE);
  w.dst   = alloci((size_t)cE);
  w.kidx  = alloci((size_t)cE * 4);

  copy_f<<<1024, 256, 0, stream>>>(w.desc0, desc0_in, (long long)SD);
  copy_f<<<1024, 256, 0, stream>>>(w.desc1, desc1_in, (long long)SD);

  // NAMES = ['self','cross','self','cross']
  const int isCross[4] = {0, 1, 0, 1};
  for (int li = 0; li < 4; ++li) {
    const LayerP& L = Ls[li];
    if (isCross[li]) {
      encoder_run(stream, w.desc0, w.desc1, L, w.deltaA, nullptr, w);
      encoder_run(stream, w.desc1, w.desc0, L, w.deltaB, nullptr, w);
      add_transpose_k<<<cB * cN, 256, 0, stream>>>(w.desc0, w.deltaA);
      add_transpose_k<<<cB * cN, 256, 0, stream>>>(w.desc1, w.deltaB);
    } else {
      // desc0 path (self-attention + DGMC refinement)
      encoder_run(stream, w.desc0, w.desc0, L, w.deltaA, w.awm, w);
      self_tail_run(stream, w.desc0, kpts0, dg, m1_w, m1_b, m2_w, m2_b, gn1g, gn1b, w);
      // desc1 path
      encoder_run(stream, w.desc1, w.desc1, L, w.deltaA, w.awm, w);
      self_tail_run(stream, w.desc1, kpts1, dg, m1_w, m1_b, m2_w, m2_b, gn1g, gn1b, w);
    }
  }

  float* out = (float*)d_out;
  copy_f<<<1024, 256, 0, stream>>>(out, w.desc0, (long long)SD);
  copy_f<<<1024, 256, 0, stream>>>(out + SD, w.desc1, (long long)SD);
}
